// BiMambaLayer_13761075216470
// MI455X (gfx1250) — compile-verified, matrix operand order checked by reading, not run on hardware
//
#include <hip/hip_runtime.h>
#include <hip/hip_bf16.h>

#define DIMD   512
#define DI     1024
#define TWO_DI 2048
#define DSTATE 16
#define DTRANK 32
#define NBATCH 2
#define LSEQ   2048
#define MTOT   (NBATCH * LSEQ)   // 4096 rows

typedef __attribute__((ext_vector_type(16))) __bf16 bf16x16;
typedef __attribute__((ext_vector_type(8)))  float  floatx8;
typedef int v4i __attribute__((__vector_size__(4 * sizeof(int))));

#ifndef __has_builtin
#define __has_builtin(x) 0
#endif

#if __has_builtin(__builtin_amdgcn_global_load_async_to_lds_b128)
#define HAVE_ASYNC_LDS 1
#else
#define HAVE_ASYNC_LDS 0
#endif

#define AS1 __attribute__((address_space(1)))
#define AS3 __attribute__((address_space(3)))

#if HAVE_ASYNC_LDS
__device__ __forceinline__ void async_copy16(void* lds_dst, const void* g_src) {
    // GLOBAL_LOAD_ASYNC_TO_LDS_B128: memory -> LDS directly, tracked by ASYNCcnt
    __builtin_amdgcn_global_load_async_to_lds_b128(
        (AS1 v4i*)g_src, (AS3 v4i*)lds_dst, /*offset=*/0, /*cpol=*/0);
}
__device__ __forceinline__ void wait_async0() {
#if __has_builtin(__builtin_amdgcn_s_wait_asynccnt)
    __builtin_amdgcn_s_wait_asynccnt(0);
#else
    asm volatile("s_wait_asynccnt 0x0" ::: "memory");
#endif
}
#endif

__device__ __forceinline__ unsigned short f2bf(float f) {
    union { float f; unsigned u; } v; v.f = f;
    unsigned u = v.u;
    unsigned r = u + 0x7FFFu + ((u >> 16) & 1u);   // round-to-nearest-even
    return (unsigned short)(r >> 16);
}

// ---------------------------------------------------------------- pack f32 -> bf16
__global__ void pack_bf16_kernel(const float* __restrict__ src,
                                 unsigned short* __restrict__ dst, int n) {
    int i = blockIdx.x * blockDim.x + threadIdx.x;
    if (i < n) dst[i] = f2bf(src[i]);
}

// ---------------------------------------------------------------- layernorm + fwd/rev bf16 pack
__global__ __launch_bounds__(256) void ln_pack_kernel(
        const float* __restrict__ x, const float* __restrict__ g,
        const float* __restrict__ b, unsigned short* __restrict__ hbf) {
    int row  = blockIdx.x * 8 + (threadIdx.x >> 5);   // one wave32 per row
    int lane = threadIdx.x & 31;
    const float* xr = x + (size_t)row * DIMD;
    float vals[16];
    float s = 0.f, s2 = 0.f;
    #pragma unroll
    for (int i = 0; i < 16; ++i) {
        float v = xr[lane + 32 * i];
        vals[i] = v; s += v; s2 += v * v;
    }
    #pragma unroll
    for (int m = 16; m >= 1; m >>= 1) {
        s  += __shfl_xor(s,  m, 32);
        s2 += __shfl_xor(s2, m, 32);
    }
    float mu  = s * (1.f / DIMD);
    float var = s2 * (1.f / DIMD) - mu * mu;
    float inv = rsqrtf(var + 1e-5f);
    int bb = row / LSEQ, l = row % LSEQ;
    int rrow = bb * LSEQ + (LSEQ - 1 - l);
    #pragma unroll
    for (int i = 0; i < 16; ++i) {
        int c = lane + 32 * i;
        float h = (vals[i] - mu) * inv * g[c] + b[c];
        unsigned short hv = f2bf(h);
        hbf[(size_t)row * DIMD + c] = hv;                              // dir 0 (forward)
        hbf[(size_t)MTOT * DIMD + (size_t)rrow * DIMD + c] = hv;       // dir 1 (reversed)
    }
}

// ---------------------------------------------------------------- bf16 WMMA GEMM: C[MxN] = A[MxK] * W[NxK]^T
#define BM 128
#define BN 64
#define BK 32

__global__ __launch_bounds__(256) void gemm_bf16_kernel(
        const unsigned short* __restrict__ A, const unsigned short* __restrict__ W,
        float* __restrict__ C, int M, int N, int K) {
    __shared__ uint4 tileA[(BM * BK) / 8];   // 512 x 16B = 8 KB
    __shared__ uint4 tileB[(BN * BK) / 8];   // 256 x 16B = 4 KB

    int tid  = threadIdx.x;
    int lane = tid & 31;
    int wave = tid >> 5;
    int wm   = wave & 3;          // wave row (4 x 32 = 128 M)
    int wn   = wave >> 2;         // wave col (2 x 32 =  64 N)
    int n0   = blockIdx.x * BN;
    int m0   = blockIdx.y * BM;
    int hh   = lane >> 4;         // K-half select per WMMA 16-bit layout
    int l16  = lane & 15;

    floatx8 c00 = {}, c01 = {}, c10 = {}, c11 = {};

    const uint4* Ag = (const uint4*)A;
    const uint4* Wg = (const uint4*)W;
    const int kTiles = K / BK;

    for (int kt = 0; kt < kTiles; ++kt) {
        int k0 = kt * BK;
#if HAVE_ASYNC_LDS
        // async global -> LDS staging (no VGPR round-trip), ASYNCcnt tracked
        {
            int c = tid;
            #pragma unroll
            for (int it = 0; it < 2; ++it, c += 256) {
                int row = c >> 2, kc = c & 3;
                async_copy16(&tileA[c], &Ag[((size_t)(m0 + row) * K + k0) / 8 + kc]);
            }
            int row = tid >> 2, kc = tid & 3;
            async_copy16(&tileB[tid], &Wg[((size_t)(n0 + row) * K + k0) / 8 + kc]);
        }
#else
        // fallback: synchronous staging through VGPRs
        {
            int c = tid;
            #pragma unroll
            for (int it = 0; it < 2; ++it, c += 256) {
                int row = c >> 2, kc = c & 3;
                tileA[c] = Ag[((size_t)(m0 + row) * K + k0) / 8 + kc];
            }
        }
        {
            int row = tid >> 2, kc = tid & 3;
            tileB[tid] = Wg[((size_t)(n0 + row) * K + k0) / 8 + kc];
        }
#endif
        if (kt + 1 < kTiles) {   // gfx1250 global_prefetch_b8 on next K tile
            __builtin_prefetch(A + (size_t)(m0 + (tid >> 2)) * K + k0 + BK, 0, 1);
            __builtin_prefetch(W + (size_t)(n0 + (tid >> 2)) * K + k0 + BK, 0, 1);
        }
#if HAVE_ASYNC_LDS
        wait_async0();
#endif
        __syncthreads();

        int ra0 = wm * 32 + l16, ra1 = ra0 + 16;
        int rb0 = wn * 32 + l16, rb1 = rb0 + 16;
        alignas(32) uint4 qa0[2], qa1[2], qb0[2], qb1[2];
        qa0[0] = tileA[ra0 * 4 + hh]; qa0[1] = tileA[ra0 * 4 + 2 + hh];
        qa1[0] = tileA[ra1 * 4 + hh]; qa1[1] = tileA[ra1 * 4 + 2 + hh];
        qb0[0] = tileB[rb0 * 4 + hh]; qb0[1] = tileB[rb0 * 4 + 2 + hh];
        qb1[0] = tileB[rb1 * 4 + hh]; qb1[1] = tileB[rb1 * 4 + 2 + hh];
        bf16x16 a0 = *reinterpret_cast<const bf16x16*>(qa0);
        bf16x16 a1 = *reinterpret_cast<const bf16x16*>(qa1);
        bf16x16 b0 = *reinterpret_cast<const bf16x16*>(qb0);
        bf16x16 b1 = *reinterpret_cast<const bf16x16*>(qb1);

        c00 = __builtin_amdgcn_wmma_f32_16x16x32_bf16(false, a0, false, b0, (short)0, c00, false, false);
        c01 = __builtin_amdgcn_wmma_f32_16x16x32_bf16(false, a0, false, b1, (short)0, c01, false, false);
        c10 = __builtin_amdgcn_wmma_f32_16x16x32_bf16(false, a1, false, b0, (short)0, c10, false, false);
        c11 = __builtin_amdgcn_wmma_f32_16x16x32_bf16(false, a1, false, b1, (short)0, c11, false, false);
        __syncthreads();
    }

    // store: VGPR r holds M = r + 8*(lane/16); N = lane%16
    int mb = m0 + wm * 32;
    int nb = n0 + wn * 32;
    #pragma unroll
    for (int r = 0; r < 8; ++r) {
        int mr = mb + r + 8 * hh;
        int nc = nb + l16;
        C[(size_t)mr * N + nc]             = c00[r];
        C[(size_t)mr * N + nc + 16]        = c01[r];
        C[(size_t)(mr + 16) * N + nc]      = c10[r];
        C[(size_t)(mr + 16) * N + nc + 16] = c11[r];
    }
}

// ---------------------------------------------------------------- depthwise causal conv(4) + bias + SiLU
__global__ void conv_silu_kernel(const float* __restrict__ xz,
                                 const float* __restrict__ conv_w,
                                 const float* __restrict__ conv_b,
                                 float* __restrict__ xc_act,
                                 unsigned short* __restrict__ xc_bf) {
    size_t i = (size_t)blockIdx.x * blockDim.x + threadIdx.x;   // over 2*M*DI
    if (i >= (size_t)2 * MTOT * DI) return;
    int d = (int)(i % DI);
    size_t md = i / DI;
    int m = (int)(md % MTOT);
    int dir = (int)(md / MTOT);
    int l = m % LSEQ;
    const float* w = conv_w + ((size_t)dir * DI + d) * 4;
    float acc = conv_b[dir * DI + d];
    #pragma unroll
    for (int k = 0; k < 4; ++k) {
        int ls = l + k - 3;
        if (ls >= 0)
            acc += w[k] * xz[((size_t)dir * MTOT + (m + k - 3)) * TWO_DI + d];
    }
    float s = acc / (1.f + __expf(-acc));   // SiLU
    xc_act[i] = s;
    xc_bf[i]  = f2bf(s);
}

// ---------------------------------------------------------------- extract dt_rank slice -> bf16
__global__ void dtr_pack_kernel(const float* __restrict__ dbl,
                                unsigned short* __restrict__ dtr_bf) {
    int i = blockIdx.x * blockDim.x + threadIdx.x;   // 2*M*32
    if (i >= 2 * MTOT * DTRANK) return;
    int r = i % DTRANK;
    int m2 = i / DTRANK;
    dtr_bf[i] = f2bf(dbl[(size_t)m2 * (DTRANK + 2 * DSTATE) + r]);
}

// ---------------------------------------------------------------- dt = softplus(dtraw + bias), in place
__global__ void softplus_kernel(float* __restrict__ dt, const float* __restrict__ dt_bias) {
    size_t i = (size_t)blockIdx.x * blockDim.x + threadIdx.x;   // 2*M*DI
    if (i >= (size_t)2 * MTOT * DI) return;
    int d = (int)(i % DI);
    int dir = (int)(i / ((size_t)MTOT * DI));
    float v = dt[i] + dt_bias[dir * DI + d];
    dt[i] = (v > 20.f) ? v : log1pf(__expf(v));
}

// ---------------------------------------------------------------- selective scan (serial over L)
__global__ __launch_bounds__(256) void scan_kernel(
        const float* __restrict__ u, const float* __restrict__ dt,
        const float* __restrict__ dbl, const float* __restrict__ A_log,
        const float* __restrict__ Dp, float* __restrict__ y) {
    int bid   = blockIdx.x;              // 16 blocks: dir(2) x batch(2) x chunk(4)
    int chunk = bid & 3;
    int bb    = (bid >> 2) & 1;
    int dir   = bid >> 3;
    int d     = chunk * 256 + threadIdx.x;

    __shared__ float sB[DSTATE], sC[DSTATE];
    float A[DSTATE], hst[DSTATE];
    #pragma unroll
    for (int n = 0; n < DSTATE; ++n) {
        A[n]   = -__expf(A_log[((size_t)dir * DI + d) * DSTATE + n]);
        hst[n] = 0.f;
    }
    float Dv = Dp[dir * DI + d];
    size_t base_row = (size_t)dir * MTOT + (size_t)bb * LSEQ;
    size_t base_md  = base_row * DI + d;

    for (int l = 0; l < LSEQ; ++l) {
        if (threadIdx.x < 32) {
            const float* r = dbl + (base_row + l) * (DTRANK + 2 * DSTATE);
            if (threadIdx.x < 16) sB[threadIdx.x]      = r[DTRANK + threadIdx.x];
            else                  sC[threadIdx.x - 16] = r[DTRANK + threadIdx.x];
        }
        __syncthreads();
        size_t idx = base_md + (size_t)l * DI;
        float dtv = dt[idx];
        float uv  = u[idx];
        float acc = uv * Dv;
        #pragma unroll
        for (int n = 0; n < DSTATE; ++n) {
            float dA = __expf(dtv * A[n]);
            hst[n] = dA * hst[n] + (dtv * sB[n]) * uv;
            acc += hst[n] * sC[n];
        }
        y[idx] = acc;
        __syncthreads();
    }
}

// ---------------------------------------------------------------- y * silu(z) -> bf16
__global__ void gate_pack_kernel(const float* __restrict__ y, const float* __restrict__ xz,
                                 unsigned short* __restrict__ ybf) {
    size_t i = (size_t)blockIdx.x * blockDim.x + threadIdx.x;   // 2*M*DI
    if (i >= (size_t)2 * MTOT * DI) return;
    int d = (int)(i % DI);
    size_t md = i / DI;
    float z = xz[md * TWO_DI + DI + d];
    float g = z / (1.f + __expf(-z));
    ybf[i] = f2bf(y[i] * g);
}

// ---------------------------------------------------------------- concat fwd + re-reversed bwd -> bf16
__global__ void concat_pack_kernel(const float* __restrict__ y0, const float* __restrict__ y1,
                                   unsigned short* __restrict__ fA) {
    size_t i = (size_t)blockIdx.x * blockDim.x + threadIdx.x;   // M*1024
    if (i >= (size_t)MTOT * 1024) return;
    int c = (int)(i % 1024);
    int m = (int)(i / 1024);
    float v;
    if (c < DIMD) v = y0[(size_t)m * DIMD + c];
    else {
        int bb = m / LSEQ, l = m % LSEQ;
        int mr = bb * LSEQ + (LSEQ - 1 - l);
        v = y1[(size_t)mr * DIMD + (c - DIMD)];
    }
    fA[i] = f2bf(v);
}

// ---------------------------------------------------------------- residual + bias
__global__ void final_kernel(const float* __restrict__ x, const float* __restrict__ fc,
                             const float* __restrict__ fb, float* __restrict__ out) {
    size_t i = (size_t)blockIdx.x * blockDim.x + threadIdx.x;   // M*512
    if (i >= (size_t)MTOT * DIMD) return;
    out[i] = x[i] + fc[i] + fb[i % DIMD];
}

// ================================================================ host launcher
extern "C" void kernel_launch(void* const* d_in, const int* in_sizes, int n_in,
                              void* d_out, int out_size, void* d_ws, size_t ws_size,
                              hipStream_t stream) {
    const float* x        = (const float*)d_in[0];
    const float* ln_g     = (const float*)d_in[1];
    const float* ln_b     = (const float*)d_in[2];
    const float* in_w     = (const float*)d_in[3];
    const float* conv_w   = (const float*)d_in[4];
    const float* conv_b   = (const float*)d_in[5];
    const float* xproj_w  = (const float*)d_in[6];
    const float* dtproj_w = (const float*)d_in[7];
    const float* dt_bias  = (const float*)d_in[8];
    const float* A_log    = (const float*)d_in[9];
    const float* Dp       = (const float*)d_in[10];
    const float* out_w    = (const float*)d_in[11];
    const float* fuse_w   = (const float*)d_in[12];
    const float* fuse_b   = (const float*)d_in[13];
    float* out = (float*)d_out;

    char* ws = (char*)d_ws;
    size_t off = 0;
    auto alloc = [&](size_t bytes) -> char* {
        char* p = ws + off;
        off = (off + bytes + 255) & ~(size_t)255;
        return p;
    };

    unsigned short* hbf    = (unsigned short*)alloc((size_t)2 * MTOT * DIMD * 2);
    unsigned short* win    = (unsigned short*)alloc((size_t)2 * TWO_DI * DIMD * 2);
    unsigned short* wxp    = (unsigned short*)alloc((size_t)2 * 64 * DI * 2);
    unsigned short* wdt    = (unsigned short*)alloc((size_t)2 * DI * DTRANK * 2);
    unsigned short* wout   = (unsigned short*)alloc((size_t)2 * DIMD * DI * 2);
    unsigned short* wfuse  = (unsigned short*)alloc((size_t)DIMD * 1024 * 2);
    float*          xz     = (float*)alloc((size_t)2 * MTOT * TWO_DI * 4);
    float*          xc_act = (float*)alloc((size_t)2 * MTOT * DI * 4);
    unsigned short* xc_bf  = (unsigned short*)alloc((size_t)2 * MTOT * DI * 2);
    float*          dbl    = (float*)alloc((size_t)2 * MTOT * 64 * 4);
    unsigned short* dtrbf  = (unsigned short*)alloc((size_t)2 * MTOT * DTRANK * 2);
    float*          dtbuf  = (float*)alloc((size_t)2 * MTOT * DI * 4);
    float*          yscan  = (float*)alloc((size_t)2 * MTOT * DI * 4);
    unsigned short* ybf    = (unsigned short*)alloc((size_t)2 * MTOT * DI * 2);
    float*          yout   = (float*)alloc((size_t)2 * MTOT * DIMD * 4);
    unsigned short* fAbf   = (unsigned short*)alloc((size_t)MTOT * 1024 * 2);
    float*          fuseC  = (float*)alloc((size_t)MTOT * DIMD * 4);

    // 1. weights -> bf16
    {
        int n;
        n = 2 * TWO_DI * DIMD;  pack_bf16_kernel<<<(n + 255) / 256, 256, 0, stream>>>(in_w, win, n);
        n = 2 * 64 * DI;        pack_bf16_kernel<<<(n + 255) / 256, 256, 0, stream>>>(xproj_w, wxp, n);
        n = 2 * DI * DTRANK;    pack_bf16_kernel<<<(n + 255) / 256, 256, 0, stream>>>(dtproj_w, wdt, n);
        n = 2 * DIMD * DI;      pack_bf16_kernel<<<(n + 255) / 256, 256, 0, stream>>>(out_w, wout, n);
        n = DIMD * 1024;        pack_bf16_kernel<<<(n + 255) / 256, 256, 0, stream>>>(fuse_w, wfuse, n);
    }

    // 2. layernorm + fwd/rev bf16 pack
    ln_pack_kernel<<<MTOT / 8, 256, 0, stream>>>(x, ln_g, ln_b, hbf);

    // 3. in-projection GEMM per dir: [4096 x 512] x [2048 x 512]^T -> [4096 x 2048]
    for (int dir = 0; dir < 2; ++dir) {
        dim3 grid(TWO_DI / BN, MTOT / BM);
        gemm_bf16_kernel<<<grid, 256, 0, stream>>>(
            hbf + (size_t)dir * MTOT * DIMD, win + (size_t)dir * TWO_DI * DIMD,
            xz + (size_t)dir * MTOT * TWO_DI, MTOT, TWO_DI, DIMD);
    }

    // 4. depthwise causal conv + SiLU
    {
        size_t n = (size_t)2 * MTOT * DI;
        conv_silu_kernel<<<(unsigned)((n + 255) / 256), 256, 0, stream>>>(xz, conv_w, conv_b, xc_act, xc_bf);
    }

    // 5. x-projection GEMM per dir: [4096 x 1024] x [64 x 1024]^T -> [4096 x 64]
    for (int dir = 0; dir < 2; ++dir) {
        dim3 grid(64 / BN, MTOT / BM);
        gemm_bf16_kernel<<<grid, 256, 0, stream>>>(
            xc_bf + (size_t)dir * MTOT * DI, wxp + (size_t)dir * 64 * DI,
            dbl + (size_t)dir * MTOT * 64, MTOT, 64, DI);
    }

    // 6. dt_rank slice -> bf16
    {
        int n = 2 * MTOT * DTRANK;
        dtr_pack_kernel<<<(n + 255) / 256, 256, 0, stream>>>(dbl, dtrbf);
    }

    // 7. dt projection GEMM per dir: [4096 x 32] x [1024 x 32]^T -> [4096 x 1024]
    for (int dir = 0; dir < 2; ++dir) {
        dim3 grid(DI / BN, MTOT / BM);
        gemm_bf16_kernel<<<grid, 256, 0, stream>>>(
            dtrbf + (size_t)dir * MTOT * DTRANK, wdt + (size_t)dir * DI * DTRANK,
            dtbuf + (size_t)dir * MTOT * DI, MTOT, DI, DTRANK);
    }

    // 8. softplus(dt + bias)
    {
        size_t n = (size_t)2 * MTOT * DI;
        softplus_kernel<<<(unsigned)((n + 255) / 256), 256, 0, stream>>>(dtbuf, dt_bias);
    }

    // 9. selective scan
    scan_kernel<<<16, 256, 0, stream>>>(xc_act, dtbuf, dbl, A_log, Dp, yscan);

    // 10. gate with silu(z), pack bf16
    {
        size_t n = (size_t)2 * MTOT * DI;
        gate_pack_kernel<<<(unsigned)((n + 255) / 256), 256, 0, stream>>>(yscan, xz, ybf);
    }

    // 11. out-projection GEMM per dir: [4096 x 1024] x [512 x 1024]^T -> [4096 x 512]
    for (int dir = 0; dir < 2; ++dir) {
        dim3 grid(DIMD / BN, MTOT / BM);
        gemm_bf16_kernel<<<grid, 256, 0, stream>>>(
            ybf + (size_t)dir * MTOT * DI, wout + (size_t)dir * DIMD * DI,
            yout + (size_t)dir * MTOT * DIMD, MTOT, DIMD, DI);
    }

    // 12. concat fwd + re-reversed bwd -> bf16 A matrix
    {
        size_t n = (size_t)MTOT * 1024;
        concat_pack_kernel<<<(unsigned)((n + 255) / 256), 256, 0, stream>>>(
            yout, yout + (size_t)MTOT * DIMD, fAbf);
    }

    // 13. fuse GEMM: [4096 x 1024] x [512 x 1024]^T -> [4096 x 512]
    {
        dim3 grid(DIMD / BN, MTOT / BM);
        gemm_bf16_kernel<<<grid, 256, 0, stream>>>(fAbf, wfuse, fuseC, MTOT, DIMD, 1024);
    }

    // 14. residual + fuse bias
    {
        size_t n = (size_t)MTOT * DIMD;
        final_kernel<<<(unsigned)((n + 255) / 256), 256, 0, stream>>>(x, fuseC, fuse_b, out);
    }
}
